// MultiHead_SelfAttention_12369505812950
// MI455X (gfx1250) — matrix operand order repaired and checked by reading, not compiled, not run on hardware
//
#include <hip/hip_runtime.h>
#include <hip/hip_bf16.h>

#define D_MODEL 1024
#define N_HEADS 16
#define HEAD_DIM 64
#define SEQ 2048
#define BATCH 2
#define M_TOTAL (BATCH * SEQ)   // 4096 rows
#define BH      (BATCH * N_HEADS)

typedef __attribute__((ext_vector_type(16))) __bf16 v16bf;
typedef __attribute__((ext_vector_type(8)))  __bf16 v8bf;
typedef __attribute__((ext_vector_type(8)))  float  v8f;

__device__ __forceinline__ v8f wmma_bf16(v16bf a, v16bf b, v8f c) {
  // (neg_a, A, neg_b, B, c_mod, C, reuse_a, reuse_b)
  return __builtin_amdgcn_wmma_f32_16x16x32_bf16(false, a, false, b, (short)0, c,
                                                 false, false);
}

__device__ __forceinline__ v8f zero_v8f() {
  v8f z;
#pragma unroll
  for (int i = 0; i < 8; ++i) z[i] = 0.0f;
  return z;
}

// A-matrix (16x32 bf16) fragment. rowp = start of this lane's row (k=0 of frag).
// Layout: lanes 0-15 hold K 0..15; lanes 16-31 hold K 16..31 (matches B frag).
__device__ __forceinline__ v16bf load_a_frag(const __bf16* rowp, int h) {
  v8bf lo = *(const v8bf*)(rowp + h * 16);
  v8bf hi = *(const v8bf*)(rowp + h * 16 + 8);
  v16bf r;
#pragma unroll
  for (int i = 0; i < 8; ++i) { r[i] = lo[i]; r[i + 8] = hi[i]; }
  return r;
}

// B-matrix (32x16 bf16) fragment. colp = element (k = h*16) of this lane's column,
// in a layout where K is contiguous. Lanes 0-15: K 0..15; lanes 16-31: K 16..31.
__device__ __forceinline__ v16bf load_b_frag(const __bf16* colp) {
  v8bf lo = *(const v8bf*)(colp);
  v8bf hi = *(const v8bf*)(colp + 8);
  v16bf r;
#pragma unroll
  for (int i = 0; i < 8; ++i) { r[i] = lo[i]; r[i + 8] = hi[i]; }
  return r;
}

// ---------------------------------------------------------------- f32 -> bf16
__global__ __launch_bounds__(256) void cvt_f32_to_bf16(
    const float* __restrict__ src, __bf16* __restrict__ dst, int n4) {
  int i = blockIdx.x * blockDim.x + threadIdx.x;
  if (i < n4) {
    float4 v = ((const float4*)src)[i];
    dst[4 * i + 0] = (__bf16)v.x;
    dst[4 * i + 1] = (__bf16)v.y;
    dst[4 * i + 2] = (__bf16)v.z;
    dst[4 * i + 3] = (__bf16)v.w;
  }
}

// ------------------------------------------------- QKV projection (Y = X W^T + b)
// Block tile 128(M) x 64(N), 8 waves, wave tile 32x32 (2x2 WMMA accumulators).
// v_transpose==0 : out[(b*H+head)*SEQ + s][d]      (Q, K layout)
// v_transpose==1 : out[(b*H+head)*HD + d][s]       (V stored transposed)
__global__ __launch_bounds__(256) void qkv_proj_kernel(
    const __bf16* __restrict__ A, const __bf16* __restrict__ W,
    const float* __restrict__ bias, __bf16* __restrict__ out, int v_transpose) {
  const int lane = threadIdx.x & 31;
  const int wv   = threadIdx.x >> 5;
  const int n_id = lane & 15;
  const int h    = lane >> 4;
  const int mBase = blockIdx.x * 128 + (wv & 3) * 32;
  const int nBase = blockIdx.y * 64 + (wv >> 2) * 32;

  v8f acc[2][2];
#pragma unroll
  for (int i = 0; i < 2; ++i)
#pragma unroll
    for (int j = 0; j < 2; ++j) acc[i][j] = zero_v8f();

  const __bf16* a0p = A + (size_t)(mBase + n_id) * D_MODEL;
  const __bf16* a1p = A + (size_t)(mBase + 16 + n_id) * D_MODEL;
  const __bf16* b0p = W + (size_t)(nBase + n_id) * D_MODEL + h * 16;
  const __bf16* b1p = W + (size_t)(nBase + 16 + n_id) * D_MODEL + h * 16;

  for (int K0 = 0; K0 < D_MODEL; K0 += 32) {
    if (K0 + 64 < D_MODEL) {
      __builtin_prefetch(a0p + K0 + 64, 0, 0);   // global_prefetch_b8
      __builtin_prefetch(b0p + K0 + 64, 0, 0);
    }
    v16bf a0 = load_a_frag(a0p + K0, h);
    v16bf a1 = load_a_frag(a1p + K0, h);
    v16bf b0 = load_b_frag(b0p + K0);
    v16bf b1 = load_b_frag(b1p + K0);
    acc[0][0] = wmma_bf16(a0, b0, acc[0][0]);
    acc[0][1] = wmma_bf16(a0, b1, acc[0][1]);
    acc[1][0] = wmma_bf16(a1, b0, acc[1][0]);
    acc[1][1] = wmma_bf16(a1, b1, acc[1][1]);
  }

#pragma unroll
  for (int j = 0; j < 2; ++j) {
    int col = nBase + j * 16 + n_id;
    float bv = bias[col];
    int head = col >> 6, d = col & 63;
#pragma unroll
    for (int i = 0; i < 2; ++i) {
#pragma unroll
      for (int e = 0; e < 8; ++e) {
        int row = mBase + i * 16 + h * 8 + e;      // C frag: m = h*8 + e
        int b = row >> 11, s = row & (SEQ - 1);
        float val = acc[i][j][e] + bv;
        size_t idx = v_transpose
            ? ((size_t)((b * N_HEADS + head) * HEAD_DIM + d)) * SEQ + s
            : ((size_t)((b * N_HEADS + head) * SEQ + s)) * HEAD_DIM + d;
        out[idx] = (__bf16)val;
      }
    }
  }
}

// ----------------------------------------------------------- flash attention
// Grid (SEQ/64, BH), 128 threads (4 waves). Each wave owns a 16-row Q strip.
__global__ __launch_bounds__(128) void attention_kernel(
    const __bf16* __restrict__ Q, const __bf16* __restrict__ K,
    const __bf16* __restrict__ Vt, __bf16* __restrict__ ctx) {
  __shared__ float p_lds[4][16][68];   // per-wave 16x64 P tile (+4 pad)
  const int lane = threadIdx.x & 31;
  const int w    = threadIdx.x >> 5;
  const int n_id = lane & 15;
  const int h    = lane >> 4;
  const int bh   = blockIdx.y;
  const int b    = bh >> 4;
  const int head = bh & 15;
  const int srow0 = blockIdx.x * 64 + w * 16;

  const __bf16* qp = Q + ((size_t)bh * SEQ + srow0 + n_id) * HEAD_DIM;
  v16bf qa0 = load_a_frag(qp, h);        // k (=d) 0..31
  v16bf qa1 = load_a_frag(qp + 32, h);   // k (=d) 32..63

  float m_r[8], l_r[8];
#pragma unroll
  for (int j = 0; j < 8; ++j) { m_r[j] = -1e30f; l_r[j] = 0.0f; }
  v8f o[4];
#pragma unroll
  for (int t = 0; t < 4; ++t) o[t] = zero_v8f();

  for (int key0 = 0; key0 < SEQ; key0 += 64) {
    // S = Q K^T for 64 keys: 4 n-tiles x 2 k-steps
    v8f s[4];
#pragma unroll
    for (int t = 0; t < 4; ++t) {
      const __bf16* kp =
          K + ((size_t)bh * SEQ + key0 + t * 16 + n_id) * HEAD_DIM + h * 16;
      v16bf kb0 = load_b_frag(kp);
      v16bf kb1 = load_b_frag(kp + 32);
      s[t] = wmma_bf16(qa0, kb0, zero_v8f());
      s[t] = wmma_bf16(qa1, kb1, s[t]);
    }
    // online softmax: rows of this strip = h*8 + j, columns across 16 lanes x 4 tiles
#pragma unroll
    for (int j = 0; j < 8; ++j) {
      float pm = -1e30f;
#pragma unroll
      for (int t = 0; t < 4; ++t) { s[t][j] *= 0.125f; pm = fmaxf(pm, s[t][j]); }
#pragma unroll
      for (int off = 8; off >= 1; off >>= 1)
        pm = fmaxf(pm, __shfl_xor(pm, off, 32));
      float mnew = fmaxf(m_r[j], pm);
      float alpha = __expf(m_r[j] - mnew);
      float ps = 0.0f;
#pragma unroll
      for (int t = 0; t < 4; ++t) {
        float e = __expf(s[t][j] - mnew);
        s[t][j] = e;
        ps += e;
      }
#pragma unroll
      for (int off = 8; off >= 1; off >>= 1) ps += __shfl_xor(ps, off, 32);
      l_r[j] = l_r[j] * alpha + ps;
      m_r[j] = mnew;
#pragma unroll
      for (int t = 0; t < 4; ++t) o[t][j] *= alpha;
    }
    // re-shape P from C-fragment layout to A-fragment layout via LDS
#pragma unroll
    for (int t = 0; t < 4; ++t)
#pragma unroll
      for (int j = 0; j < 8; ++j)
        p_lds[w][h * 8 + j][t * 16 + n_id] = s[t][j];
    __syncthreads();
    v16bf pa0, pa1;
#pragma unroll
    for (int i = 0; i < 16; ++i) {
      int k0 = h * 16 + i;
      pa0[i] = (__bf16)p_lds[w][n_id][k0];
      pa1[i] = (__bf16)p_lds[w][n_id][32 + k0];
    }
    __syncthreads();
    // O += P V : V^T layout makes B-fragment K-contiguous
#pragma unroll
    for (int t = 0; t < 4; ++t) {
      const __bf16* vp =
          Vt + ((size_t)bh * HEAD_DIM + t * 16 + n_id) * SEQ + key0 + h * 16;
      v16bf vb0 = load_b_frag(vp);
      v16bf vb1 = load_b_frag(vp + 32);
      o[t] = wmma_bf16(pa0, vb0, o[t]);
      o[t] = wmma_bf16(pa1, vb1, o[t]);
    }
  }
  // normalize and store ctx as [B,S,D] bf16 for the output projection
#pragma unroll
  for (int j = 0; j < 8; ++j) {
    float inv = 1.0f / l_r[j];
    int srow = srow0 + h * 8 + j;
#pragma unroll
    for (int t = 0; t < 4; ++t) {
      int d = t * 16 + n_id;
      ctx[((size_t)(b * SEQ + srow)) * D_MODEL + head * HEAD_DIM + d] =
          (__bf16)(o[t][j] * inv);
    }
  }
}

// ------------------------------------- output projection + bias + residual (f32)
__global__ __launch_bounds__(256) void out_proj_kernel(
    const __bf16* __restrict__ A, const __bf16* __restrict__ W,
    const float* __restrict__ bias, const float* __restrict__ x,
    float* __restrict__ res) {
  const int lane = threadIdx.x & 31;
  const int wv   = threadIdx.x >> 5;
  const int n_id = lane & 15;
  const int h    = lane >> 4;
  const int mBase = blockIdx.x * 128 + (wv & 3) * 32;
  const int nBase = blockIdx.y * 64 + (wv >> 2) * 32;

  v8f acc[2][2];
#pragma unroll
  for (int i = 0; i < 2; ++i)
#pragma unroll
    for (int j = 0; j < 2; ++j) acc[i][j] = zero_v8f();

  const __bf16* a0p = A + (size_t)(mBase + n_id) * D_MODEL;
  const __bf16* a1p = A + (size_t)(mBase + 16 + n_id) * D_MODEL;
  const __bf16* b0p = W + (size_t)(nBase + n_id) * D_MODEL + h * 16;
  const __bf16* b1p = W + (size_t)(nBase + 16 + n_id) * D_MODEL + h * 16;

  for (int K0 = 0; K0 < D_MODEL; K0 += 32) {
    if (K0 + 64 < D_MODEL) {
      __builtin_prefetch(a0p + K0 + 64, 0, 0);
      __builtin_prefetch(b0p + K0 + 64, 0, 0);
    }
    v16bf a0 = load_a_frag(a0p + K0, h);
    v16bf a1 = load_a_frag(a1p + K0, h);
    v16bf b0 = load_b_frag(b0p + K0);
    v16bf b1 = load_b_frag(b1p + K0);
    acc[0][0] = wmma_bf16(a0, b0, acc[0][0]);
    acc[0][1] = wmma_bf16(a0, b1, acc[0][1]);
    acc[1][0] = wmma_bf16(a1, b0, acc[1][0]);
    acc[1][1] = wmma_bf16(a1, b1, acc[1][1]);
  }

#pragma unroll
  for (int j = 0; j < 2; ++j) {
    int col = nBase + j * 16 + n_id;
    float bv = bias[col];
#pragma unroll
    for (int i = 0; i < 2; ++i) {
#pragma unroll
      for (int e = 0; e < 8; ++e) {
        int row = mBase + i * 16 + h * 8 + e;
        size_t idx = (size_t)row * D_MODEL + col;
        res[idx] = acc[i][j][e] + bv + x[idx];
      }
    }
  }
}

// ------------------------------------------------------------------ LayerNorm
__global__ __launch_bounds__(256) void layernorm_kernel(
    const float* __restrict__ res, const float* __restrict__ gamma,
    const float* __restrict__ beta, float* __restrict__ out) {
  __shared__ float red[256];
  const int row = blockIdx.x;
  const float* r = res + (size_t)row * D_MODEL;
  float v[4];
  float s = 0.0f;
#pragma unroll
  for (int i = 0; i < 4; ++i) {
    v[i] = r[threadIdx.x + i * 256];
    s += v[i];
  }
  red[threadIdx.x] = s;
  __syncthreads();
  for (int st = 128; st > 0; st >>= 1) {
    if (threadIdx.x < st) red[threadIdx.x] += red[threadIdx.x + st];
    __syncthreads();
  }
  float mean = red[0] * (1.0f / D_MODEL);
  __syncthreads();
  float s2 = 0.0f;
#pragma unroll
  for (int i = 0; i < 4; ++i) {
    float d = v[i] - mean;
    s2 += d * d;
  }
  red[threadIdx.x] = s2;
  __syncthreads();
  for (int st = 128; st > 0; st >>= 1) {
    if (threadIdx.x < st) red[threadIdx.x] += red[threadIdx.x + st];
    __syncthreads();
  }
  float rstd = rsqrtf(red[0] * (1.0f / D_MODEL) + 1e-6f);
#pragma unroll
  for (int i = 0; i < 4; ++i) {
    int c = threadIdx.x + i * 256;
    out[(size_t)row * D_MODEL + c] = (v[i] - mean) * rstd * gamma[c] + beta[c];
  }
}

extern "C" void kernel_launch(void* const* d_in, const int* in_sizes, int n_in,
                              void* d_out, int out_size, void* d_ws, size_t ws_size,
                              hipStream_t stream) {
  const float* x     = (const float*)d_in[0];
  const float* Wq    = (const float*)d_in[1];
  const float* bq    = (const float*)d_in[2];
  const float* Wk    = (const float*)d_in[3];
  const float* bk    = (const float*)d_in[4];
  const float* Wv    = (const float*)d_in[5];
  const float* bv    = (const float*)d_in[6];
  const float* Wo    = (const float*)d_in[7];
  const float* bo    = (const float*)d_in[8];
  const float* gamma = (const float*)d_in[9];
  const float* beta  = (const float*)d_in[10];

  char* ws = (char*)d_ws;
  size_t off = 0;
  __bf16* xb  = (__bf16*)(ws + off); off += (size_t)M_TOTAL * D_MODEL * 2;
  __bf16* wqb = (__bf16*)(ws + off); off += (size_t)D_MODEL * D_MODEL * 2;
  __bf16* wkb = (__bf16*)(ws + off); off += (size_t)D_MODEL * D_MODEL * 2;
  __bf16* wvb = (__bf16*)(ws + off); off += (size_t)D_MODEL * D_MODEL * 2;
  __bf16* wob = (__bf16*)(ws + off); off += (size_t)D_MODEL * D_MODEL * 2;
  __bf16* Qb  = (__bf16*)(ws + off); off += (size_t)BH * SEQ * HEAD_DIM * 2;
  __bf16* Kb  = (__bf16*)(ws + off); off += (size_t)BH * SEQ * HEAD_DIM * 2;
  __bf16* Vtb = (__bf16*)(ws + off); off += (size_t)BH * SEQ * HEAD_DIM * 2;
  __bf16* ctxb = (__bf16*)(ws + off); off += (size_t)M_TOTAL * D_MODEL * 2;
  float* resf  = (float*)(ws + off);  off += (size_t)M_TOTAL * D_MODEL * 4;

  const int nX4 = M_TOTAL * D_MODEL / 4;   // 1048576
  const int nW4 = D_MODEL * D_MODEL / 4;   // 262144
  cvt_f32_to_bf16<<<(nX4 + 255) / 256, 256, 0, stream>>>(x, xb, nX4);
  cvt_f32_to_bf16<<<(nW4 + 255) / 256, 256, 0, stream>>>(Wq, wqb, nW4);
  cvt_f32_to_bf16<<<(nW4 + 255) / 256, 256, 0, stream>>>(Wk, wkb, nW4);
  cvt_f32_to_bf16<<<(nW4 + 255) / 256, 256, 0, stream>>>(Wv, wvb, nW4);
  cvt_f32_to_bf16<<<(nW4 + 255) / 256, 256, 0, stream>>>(Wo, wob, nW4);

  dim3 gproj(M_TOTAL / 128, D_MODEL / 64);  // 32 x 16
  qkv_proj_kernel<<<gproj, 256, 0, stream>>>(xb, wqb, bq, Qb, 0);
  qkv_proj_kernel<<<gproj, 256, 0, stream>>>(xb, wkb, bk, Kb, 0);
  qkv_proj_kernel<<<gproj, 256, 0, stream>>>(xb, wvb, bv, Vtb, 1);

  attention_kernel<<<dim3(SEQ / 64, BH), 128, 0, stream>>>(Qb, Kb, Vtb, ctxb);

  out_proj_kernel<<<gproj, 256, 0, stream>>>(ctxb, wob, bo, x, resf);

  layernorm_kernel<<<M_TOTAL, 256, 0, stream>>>(resf, gamma, beta, (float*)d_out);
}
